// TemporalGCN_43035572306084
// MI455X (gfx1250) — matrix operand order repaired and checked by reading, not compiled, not run on hardware
//
#include <hip/hip_runtime.h>
#include <hip/hip_bf16.h>
#include <math.h>

// ---- problem constants (match reference) ----
#define N_NODES 32768
#define N_EDGES 327680
#define IN_CH   128
#define HID_CH  128
#define OUT_CH  192
#define BATCH   64
#define SEQ     512
#define NHEAD   6
#define DH      32
#define GDIM    (4*OUT_CH)   // 768 LSTM gate dim

typedef __attribute__((ext_vector_type(16))) __bf16 v16bf;
typedef __attribute__((ext_vector_type(8)))  __bf16 v8bf;
typedef __attribute__((ext_vector_type(8)))  float  v8f;
typedef __attribute__((ext_vector_type(4)))  unsigned int v4u;
typedef __attribute__((ext_vector_type(8)))  int v8i;
typedef __attribute__((ext_vector_type(4)))  int v4i;

// ---------------- WMMA helper (CDNA5 16x16x32 bf16, f32 accum) ---------------
__device__ __forceinline__ v8f wmma_bf16(v16bf a, v16bf b, v8f c) {
  return __builtin_amdgcn_wmma_f32_16x16x32_bf16(false, a, false, b, (short)0, c,
                                                 false, false);
}

// stop the pre-RA scheduler from hoisting loads across this point (reg pressure)
__device__ __forceinline__ void sched_fence() {
#if __has_builtin(__builtin_amdgcn_sched_barrier)
  __builtin_amdgcn_sched_barrier(0);
#else
  asm volatile("" ::: "memory");
#endif
}

// ---- swizzled tile layouts: tile = 512 bf16, lane L owns bytes [L*32, L*32+32)
// A-tile (16 rows x 32 k): lane = row + 16*(kk>=16);
// elem = kk & 15
__device__ __forceinline__ int a_off(int row, int kk) {
  return (row + 16 * (kk >> 4)) * 16 + (kk & 15);
}
// B-tile (32 k x 16 cols): lane = col + 16*(k>=16); elem = k & 15
__device__ __forceinline__ int b_off(int k, int col) {
  return (col + 16 * (k >> 4)) * 16 + (k & 15);
}

// fragment load from a swizzled tile: 2x 16-byte vector loads (b128)
__device__ __forceinline__ v16bf load_frag_sw(const __bf16* __restrict__ base) {
  const int lane = threadIdx.x & 31;
  const v8bf* p = (const v8bf*)(base + lane * 16);
  union { v16bf v; v8bf h[2]; } u;
  u.h[0] = p[0];
  u.h[1] = p[1];
  return u.v;
}

__device__ __forceinline__ unsigned pack_bf16(float x, float y) {
  union { __bf16 b[2]; unsigned u; } p;
  p.b[0] = (__bf16)x; p.b[1] = (__bf16)y;
  return p.u;
}

// ---------------- generic tiled WMMA GEMM: C[M,N] = A[M,K] @ B (+bias,relu) ---
// Block = 8 waves, tile 64x64, k-step 32. LDS tiles stored pre-swizzled in the
// WMMA fragment layout so each wave issues b128 fragment loads.
template <int TRANSB>
__global__ __launch_bounds__(256) void gemm_wmma(
    const float* __restrict__ A, const float* __restrict__ B,
    const float* __restrict__ bias, float* __restrict__ C,
    int M, int N, int K, int relu)
{
  __shared__ __align__(16) __bf16 As_sw[4 * 512];   // 4 m-tiles
  __shared__ __align__(16) __bf16 Bs_sw[4 * 512];   // 4 n-tiles
  const int tid = threadIdx.x;
  const int wid = tid >> 5, lane = tid & 31;
  const int wy = wid >> 1, wx = wid & 1;            // 4x2 wave grid
  const int bm = blockIdx.y * 64, bn = blockIdx.x * 64;
  v8f acc0 = {}; v8f acc1 = {};

  for (int k0 = 0; k0 < K; k0 += 32) {
    __syncthreads();
#pragma unroll
    for (int i = 0; i < 4; ++i) {                   // 1024 packed A pairs
      int p = tid * 4 + i;
      int r = p >> 4, c = (p & 15) * 2;
      float2 a2 = *(const float2*)&A[(size_t)(bm + r) * K + k0 + c];
      *(unsigned*)&As_sw[(r >> 4) * 512 + a_off(r & 15, c)] = pack_bf16(a2.x, a2.y);
    }
#pragma unroll
    for (int i = 0; i < 4; ++i) {                   // 1024 packed B pairs
      int p = tid * 4 + i;
      int k = (p >> 6) * 2, c = p & 63;
      float x, y;
      if (TRANSB) {
        float2 b2 = *(const float2*)&B[(size_t)(bn + c) * K + k0 + k];
        x = b2.x; y = b2.y;
      } else {
        x = B[(size_t)(k0 + k) * N + bn + c];
        y = B[(size_t)(k0 + k + 1) * N + bn + c];
      }
      *(unsigned*)&Bs_sw[(c >> 4) * 512 + b_off(k, c & 15)] = pack_bf16(x, y);
    }
    if (k0 + 32 < K)                                // global_prefetch_b8 path
      __builtin_prefetch(&A[(size_t)(bm + (tid >> 2)) * K + k0 + 32], 0, 1);
    __syncthreads();
    v16bf a  = load_frag_sw(&As_sw[wy * 512]);
    v16bf b0 = load_frag_sw(&Bs_sw[(wx * 2 + 0) * 512]);
    v16bf b1 = load_frag_sw(&Bs_sw[(wx * 2 + 1) * 512]);
    acc0 = wmma_bf16(a, b0, acc0);
    acc1 = wmma_bf16(a, b1, acc1);
  }

  const int col = lane & 15, half = lane >> 4;
  const int n0 = bn + wx * 32 + col;
  const float bi0 = bias ? bias[n0] : 0.f;
  const float bi1 = bias ? bias[n0 + 16] : 0.f;
#pragma unroll
  for (int r = 0; r < 8; ++r) {
    int m = bm + wy * 16 + r + 8 * half;
    float v0 = acc0[r] + bi0;
    float v1 = acc1[r] + bi1;
    if (relu) { v0 = fmaxf(v0, 0.f); v1 = fmaxf(v1, 0.f); }
    C[(size_t)m * N + n0]      = v0;
    C[(size_t)m * N + n0 + 16] = v1;
  }
}

// ---------------- small elementwise / graph kernels ---------------------------
__global__ void zero_f(float* p, int n) {
  int i = blockIdx.x * blockDim.x + threadIdx.x;
  if (i < n) p[i] = 0.f;
}

__global__ void edge_deg(const int* __restrict__ dst, float* __restrict__ deg, int E) {
  int i = blockIdx.x * blockDim.x + threadIdx.x;
  if (i < E) atomicAdd(&deg[dst[i]], 1.0f);
}

__global__ void dinv_k(const float* __restrict__ deg, float* __restrict__ dinv, int n) {
  int i = blockIdx.x * blockDim.x + threadIdx.x;
  if (i < n) dinv[i] = rsqrtf(deg[i] + 1.0f);      // +1 = self loop
}

// out[dst] += dinv[src]*dinv[dst] * t[src]  (per channel); C compile-time
template <int C>
__global__ void scatter_k(const float* __restrict__ t, const int* __restrict__ src,
                          const int* __restrict__ dst, const float* __restrict__ dinv,
                          float* __restrict__ out, int E)
{
  int i = blockIdx.x * blockDim.x + threadIdx.x;
  if (i >= E * C) return;
  int e = i / C, c = i - e * C;
  int s = src[e], d = dst[e];
  atomicAdd(&out[(size_t)d * C + c], dinv[s] * dinv[d] * t[(size_t)s * C + c]);
}

// out = relu(out + dinv^2 * t (self loop) + bias)
template <int C>
__global__ void bias_relu_self(float* __restrict__ out, const float* __restrict__ t,
                               const float* __restrict__ dinv,
                               const float* __restrict__ bias, int n)
{
  int i = blockIdx.x * blockDim.x + threadIdx.x;
  if (i >= n * C) return;
  int node = i / C, c = i - node * C;
  float di = dinv[node];
  out[i] = fmaxf(out[i] + di * di * t[i] + bias[c], 0.f);
}

// pre-swizzle w_hh^T into WMMA B-fragment tiles (bf16) + fuse biases.
// tile index = (ks*4 + g)*12 + j ; 6 ksteps * 4 gates * 12 j-tiles = 288 tiles.
__global__ void prep_lstm(const float* __restrict__ w_hh, const float* __restrict__ b_ih,
                          const float* __restrict__ b_hh, __bf16* __restrict__ whhT_sw,
                          float* __restrict__ bsum)
{
  int i = blockIdx.x * blockDim.x + threadIdx.x;
  if (i < GDIM) bsum[i] = b_ih[i] + b_hh[i];
  if (i < 6 * 4 * 12 * 512) {
    int tile = i >> 9, within = i & 511;
    int lane = within >> 4, e = within & 15;
    int col = lane & 15, half = lane >> 4;
    int klocal = half * 16 + e;                    // B-layout inverse
    int j = tile % 12, kg = tile / 12;
    int ks = kg >> 2, g = kg & 3;
    int kglob = ks * 32 + klocal;
    int nglob = g * OUT_CH + j * 16 + col;
    whhT_sw[i] = (__bf16)w_hh[(size_t)nglob * OUT_CH + kglob];
  }
}

// ---------------- persistent single-WGP LSTM --------------------------------
// 512 threads = 16 waves, single workgroup. CDNA5 has 320 KB LDS per WGP:
// the entire swizzled w_hh^T (288 KB) is staged into LDS once via the Tensor
// Data Mover (one TENSOR_LOAD_TO_LDS descriptor, TENSORcnt-waited), h-state
// tiles take 24 KB, so every recurrent B fragment is a ds_load_b128 from
// on-WGP SRAM. c stays in accumulator VGPRs. A laundered offset (empty asm)
// keeps LICM from hoisting 576 VGPRs of fragments out of the t-loop.
__global__ __launch_bounds__(512) void lstm_wmma(
    const float* __restrict__ xg, const __bf16* __restrict__ whhT_sw,
    float* __restrict__ hs)
{
  __shared__ __align__(16) __bf16 whh_lds[288 * 512];   // 288 KB fragment tiles
  __shared__ __align__(16) __bf16 hsh_sw[4 * 6 * 512];  // (m,ks) A-tiles, 24 KB
  const int tid = threadIdx.x, wid = tid >> 5, lane = tid & 31;
  const int col = lane & 15, half = lane >> 4;

#if __has_builtin(__builtin_amdgcn_tensor_load_to_lds) && \
    __has_builtin(__builtin_amdgcn_s_wait_tensorcnt)
  if (tid == 0) {   // one TDM descriptor: 2-D tile 512x144 dwords = 288 KB
    unsigned lds_lo = (unsigned)(size_t)(void*)&whh_lds[0];
    unsigned long long ga = (unsigned long long)(size_t)whhT_sw;
    v4u g0;
    g0[0] = 1u;                                   // count=1, user descriptor
    g0[1] = lds_lo;                               // D#.lds_addr
    g0[2] = (unsigned)(ga & 0xFFFFFFFFu);         // global_addr[31:0]
    g0[3] = (unsigned)(((ga >> 32) & 0x01FFFFFFu) | (2u << 30)); // [56:32]|type=2
    v8i g1;
    g1[0] = 0x20000;                              // data_size=2 (4B), no mask
    g1[1] = (int)(512u << 16);                    // tensor_dim0 = 512 (lo16)
    g1[2] = (int)(144u << 16);                    // dim0 hi | tensor_dim1 = 144
    g1[3] = (int)(512u << 16);                    // dim1 hi | tile_dim0 = 512
    g1[4] = 144;                                  // tile_dim1=144, tile_dim2=0
    g1[5] = 512;                                  // tensor_dim0_stride = 512
    g1[6] = 0;
    g1[7] = 0;
    v4i g2 = {0, 0, 0, 0};                        // <=2-D tensor
    v4i g3 = {0, 0, 0, 0};
    v8i g4 = {0, 0, 0, 0, 0, 0, 0, 0};            // 6-arg toolchain variant
    __builtin_amdgcn_tensor_load_to_lds(g0, g1, g2, g3, g4, 0);
    __builtin_amdgcn_s_wait_tensorcnt((short)0);  // s_wait_tensorcnt 0
  }
#else
  {  // fallback stage: global -> LDS, 16B chunks
    const uint4* s4 = (const uint4*)whhT_sw;
    uint4* d4 = (uint4*)whh_lds;
    for (int i = tid; i < (288 * 512 * 2) / 16; i += 512) d4[i] = s4[i];
  }
#endif
  for (int i = tid; i < 4 * 6 * 512; i += 512) hsh_sw[i] = (__bf16)0.f;

  float creg[3][8];
#pragma unroll
  for (int cb = 0; cb < 3; ++cb)
#pragma unroll
    for (int r = 0; r < 8; ++r) creg[cb][r] = 0.f;

  int wofs = 0;                                    // laundered: always 0
  for (int t = 0; t < SEQ; ++t) {
    asm volatile("" : "+v"(wofs));                 // defeat LICM of B-frag loads
    __syncthreads();                               // h_prev (and staged W) stable
    float hnew[3][8];
#pragma unroll
    for (int cb = 0; cb < 3; ++cb) {
      const int combo = wid + cb * 16;             // 48 (m,j) tiles / 16 waves
      const int m = combo / 12, j = combo % 12;

      v8f acc[4];
#pragma unroll
      for (int g = 0; g < 4; ++g)                  // init with x-gates (+bias)
#pragma unroll
        for (int r = 0; r < 8; ++r) {
          int b = m * 16 + r + 8 * half;
          acc[g][r] = xg[(size_t)(b * SEQ + t) * GDIM + g * OUT_CH + j * 16 + col];
        }
      sched_fence();
#pragma unroll
      for (int ks = 0; ks < 6; ++ks) {             // K = 192 = 6 k-steps
        v16bf a = load_frag_sw(&hsh_sw[(m * 6 + ks) * 512]);
#pragma unroll
        for (int g = 0; g < 4; ++g) {
          v16bf bf = load_frag_sw(&whh_lds[((ks * 4 + g) * 12 + j) * 512 + wofs]);
          acc[g] = wmma_bf16(a, bf, acc[g]);
        }
        sched_fence();                             // cap live B fragments
      }
#pragma unroll
      for (int r = 0; r < 8; ++r) {                // i,f,g,o gate math in regs
        float ig = 1.f / (1.f + __expf(-acc[0][r]));
        float fg = 1.f / (1.f + __expf(-acc[1][r]));
        float gg = tanhf(acc[2][r]);
        float og = 1.f / (1.f + __expf(-acc[3][r]));
        float c  = fg * creg[cb][r] + ig * gg;
        creg[cb][r] = c;
        hnew[cb][r] = og * tanhf(c);
      }
      sched_fence();
    }
    __syncthreads();                               // all reads of h_prev done
#pragma unroll
    for (int cb = 0; cb < 3; ++cb) {
      const int combo = wid + cb * 16;
      const int m = combo / 12, j = combo % 12;
#pragma unroll
      for (int r = 0; r < 8; ++r) {
        int b  = m * 16 + r + 8 * half;
        int hc = j * 16 + col;
        int ks2 = hc >> 5, kk = hc & 31;           // write in swizzled A layout
        hsh_sw[(m * 6 + ks2) * 512 + a_off(r + 8 * half, kk)] = (__bf16)hnew[cb][r];
        hs[(size_t)(b * SEQ + t) * OUT_CH + hc] = hnew[cb][r];
      }
    }
  }
}

// ---------------- flash attention: block = (batch, head, 128 q-rows) ---------
__global__ __launch_bounds__(256) void attn_wmma(
    const float* __restrict__ qkv, float* __restrict__ obuf)
{
  const int bi  = blockIdx.x;
  const int b   = bi / (NHEAD * 4);
  const int rem = bi % (NHEAD * 4);
  const int hd  = rem >> 2, qb = rem & 3;
  const int tid = threadIdx.x, wid = tid >> 5, lane = tid & 31;
  const int col = lane & 15, half = lane >> 4;

  __shared__ __align__(16) __bf16 Qs_sw[8 * 512];  // 8 m-tiles (swizzled A)
  __shared__ __align__(16) __bf16 Kt_sw[2 * 512];  // K^T B-tiles (k=feat)
  __shared__ __align__(16) __bf16 Vs_sw[2 * 512];  // V  B-tiles (k=kv row)
  __shared__ __align__(16) __bf16 Ps_sw[8 * 512];  // per-wave P as A-tile

  const float scale = 0.17677669529663687f;        // 1/sqrt(32)
#pragma unroll
  for (int i = 0; i < 8; ++i) {                    // 2048 Q pairs
    int p = tid * 8 + i;
    int r = p >> 4, f = (p & 15) * 2;
    int node = b * SEQ + qb * 128 + r;
    float2 q2 = *(const float2*)&qkv[(size_t)node * (3 * OUT_CH) + hd * DH + f];
    *(unsigned*)&Qs_sw[(r >> 4) * 512 + a_off(r & 15, f)] =
        pack_bf16(q2.x * scale, q2.y * scale);
  }

  v8f o0 = {}; v8f o1 = {};
  float Mrow[8], Lrow[8];
#pragma unroll
  for (int r = 0; r < 8; ++r) { Mrow[r] = -1e30f; Lrow[r] = 0.f; }

  for (int kt = 0; kt < SEQ / 32; ++kt) {
    __syncthreads();
#pragma unroll
    for (int i = 0; i < 2; ++i) {                  // 512 K pairs (pair along f)
      int p = tid * 2 + i;
      int kv = p >> 4, f = (p & 15) * 2;
      int node = b * SEQ + kt * 32 + kv;
      float2 k2 = *(const float2*)&qkv[(size_t)node * (3 * OUT_CH) + OUT_CH + hd * DH + f];
      *(unsigned*)&Kt_sw[(kv >> 4) * 512 + b_off(f, kv & 15)] = pack_bf16(k2.x, k2.y);
    }
#pragma unroll
    for (int i = 0; i < 2; ++i) {                  // 512 V pairs (pair along kv)
      int p = tid * 2 + i;
      int f = p >> 4, kv = (p & 15) * 2;
      float v0 = qkv[(size_t)(b * SEQ + kt * 32 + kv) * (3 * OUT_CH) + 2 * OUT_CH + hd * DH + f];
      float v1 = qkv[(size_t)(b * SEQ + kt * 32 + kv + 1) * (3 * OUT_CH) + 2 * OUT_CH + hd * DH + f];
      *(unsigned*)&Vs_sw[(f >> 4) * 512 + b_off(kv, f & 15)] = pack_bf16(v0, v1);
    }
    __syncthreads();

    v16bf qa = load_frag_sw(&Qs_sw[wid * 512]);
    v8f s0 = {}; v8f s1 = {};
    s0 = wmma_bf16(qa, load_frag_sw(&Kt_sw[0]),   s0);   // dh=32 = 1 k-step
    s1 = wmma_bf16(qa, load_frag_sw(&Kt_sw[512]), s1);

#pragma unroll
    for (int r = 0; r < 8; ++r) {                  // online softmax per row
      float mx = fmaxf(s0[r], s1[r]);
#pragma unroll
      for (int d = 1; d <= 8; d <<= 1) mx = fmaxf(mx, __shfl_xor(mx, d, 32));
      float newM = fmaxf(Mrow[r], mx);
      float corr = __expf(Mrow[r] - newM);
      float p0 = __expf(s0[r] - newM);
      float p1 = __expf(s1[r] - newM);
      float rs = p0 + p1;
#pragma unroll
      for (int d = 1; d <= 8; d <<= 1) rs += __shfl_xor(rs, d, 32);
      Lrow[r] = Lrow[r] * corr + rs;
      Mrow[r] = newM;
      o0[r] *= corr; o1[r] *= corr;
      // store P directly in swizzled A layout (kk = col / col+16)
      Ps_sw[wid * 512 + a_off(r + 8 * half, col)]      = (__bf16)p0;
      Ps_sw[wid * 512 + a_off(r + 8 * half, col + 16)] = (__bf16)p1;
    }
    v16bf pa = load_frag_sw(&Ps_sw[wid * 512]);
    o0 = wmma_bf16(pa, load_frag_sw(&Vs_sw[0]),   o0);
    o1 = wmma_bf16(pa, load_frag_sw(&Vs_sw[512]), o1);
  }

#pragma unroll
  for (int r = 0; r < 8; ++r) {
    float inv = 1.f / Lrow[r];
    int node = b * SEQ + qb * 128 + wid * 16 + r + 8 * half;
    obuf[(size_t)node * OUT_CH + hd * DH + col]      = o0[r] * inv;
    obuf[(size_t)node * OUT_CH + hd * DH + 16 + col] = o1[r] * inv;
  }
}

// ---------------- mean over sequence (mean commutes with out-proj) -----------
__global__ void mean_k(const float* __restrict__ obuf, float* __restrict__ mbuf) {
  int i = blockIdx.x * blockDim.x + threadIdx.x;
  if (i >= BATCH * OUT_CH) return;
  int b = i / OUT_CH, c = i - b * OUT_CH;
  float s = 0.f;
  for (int t = 0; t < SEQ; ++t) s += obuf[(size_t)(b * SEQ + t) * OUT_CH + c];
  mbuf[i] = s * (1.f / SEQ);
}

// ============================ host orchestration ============================
extern "C" void kernel_launch(void* const* d_in, const int* in_sizes, int n_in,
                              void* d_out, int out_size, void* d_ws, size_t ws_size,
                              hipStream_t stream) {
  const float* x    = (const float*)d_in[0];
  const int*   ei   = (const int*)d_in[1];
  const float* W0   = (const float*)d_in[3];
  const float* b0   = (const float*)d_in[4];
  const float* W1   = (const float*)d_in[5];
  const float* b1   = (const float*)d_in[6];
  const float* W2   = (const float*)d_in[7];
  const float* b2   = (const float*)d_in[8];
  const float* w_ih = (const float*)d_in[9];
  const float* w_hh = (const float*)d_in[10];
  const float* b_ih = (const float*)d_in[11];
  const float* b_hh = (const float*)d_in[12];
  const float* ipw  = (const float*)d_in[13];
  const float* ipb  = (const float*)d_in[14];
  const float* opw  = (const float*)d_in[15];
  const float* opb  = (const float*)d_in[16];
  const int* src = ei;
  const int* dst = ei + N_EDGES;
  float* out = (float*)d_out;

  char* ws = (char*)d_ws;
  size_t off = 0;
  auto alloc = [&](size_t nfloats) -> float* {
    float* p = (float*)(ws + off);
    off += ((nfloats * 4 + 255) / 256) * 256;
    return p;
  };
  float* deg   = alloc(N_NODES);
  float* dinv  = alloc(N_NODES);
  float* t     = alloc((size_t)N_NODES * OUT_CH);
  float* hA    = alloc((size_t)N_NODES * OUT_CH);
  float* hB    = alloc((size_t)N_NODES * OUT_CH);
  float* xg    = alloc((size_t)N_NODES * GDIM);
  float* hsb   = alloc((size_t)N_NODES * OUT_CH);
  float* qkvb  = alloc((size_t)N_NODES * 3 * OUT_CH);
  float* ob    = alloc((size_t)N_NODES * OUT_CH);
  float* mb    = alloc(BATCH * OUT_CH);
  float* bsum  = alloc(GDIM);
  __bf16* whhT = (__bf16*)alloc((GDIM * OUT_CH) / 2);

  dim3 blk(256);
  // ---- GCN normalization ----
  zero_f  <<<(N_NODES + 255) / 256, blk, 0, stream>>>(deg, N_NODES);
  edge_deg<<<(N_EDGES + 255) / 256, blk, 0, stream>>>(dst, deg, N_EDGES);
  dinv_k  <<<(N_NODES + 255) / 256, blk, 0, stream>>>(deg, dinv, N_NODES);

  auto gcn = [&](const float* hin, const float* W, const float* bias, float* hout,
                 int Cin, int Cout) {
    gemm_wmma<0><<<dim3(Cout / 64, N_NODES / 64), 256, 0, stream>>>(
        hin, W, nullptr, t, N_NODES, Cout, Cin, 0);
    int nt = N_NODES * Cout;
    zero_f<<<(nt + 255) / 256, blk, 0, stream>>>(hout, nt);
    long total = (long)N_EDGES * Cout;
    if (Cout == 128) {
      scatter_k<128><<<(int)((total + 255) / 256), blk, 0, stream>>>(
          t, src, dst, dinv, hout, N_EDGES);
      bias_relu_self<128><<<(nt + 255) / 256, blk, 0, stream>>>(
          hout, t, dinv, bias, N_NODES);
    } else {
      scatter_k<192><<<(int)((total + 255) / 256), blk, 0, stream>>>(
          t, src, dst, dinv, hout, N_EDGES);
      bias_relu_self<192><<<(nt + 255) / 256, blk, 0, stream>>>(
          hout, t, dinv, bias, N_NODES);
    }
  };
  gcn(x,  W0, b0, hA, IN_CH,  HID_CH);
  gcn(hA, W1, b1, hB, HID_CH, HID_CH);
  gcn(hB, W2, b2, hA, HID_CH, OUT_CH);            // hA = h3 [N_NODES, 192]

  // ---- LSTM: hoist x@w_ih^T into one big WMMA GEMM, persist the recurrence --
  prep_lstm<<<(GDIM * OUT_CH + 255) / 256, blk, 0, stream>>>(w_hh, b_ih, b_hh,
                                                             whhT, bsum);
  gemm_wmma<1><<<dim3(GDIM / 64, N_NODES / 64), 256, 0, stream>>>(
      hA, w_ih, bsum, xg, N_NODES, GDIM, OUT_CH, 0);
  lstm_wmma<<<1, 512, 0, stream>>>(xg, whhT, hsb);

  // ---- attention ----
  gemm_wmma<1><<<dim3((3 * OUT_CH) / 64, N_NODES / 64), 256, 0, stream>>>(
      hsb, ipw, ipb, qkvb, N_NODES, 3 * OUT_CH, OUT_CH, 0);
  attn_wmma<<<BATCH * NHEAD * 4, 256, 0, stream>>>(qkvb, ob);

  // ---- mean, then out-projection (they commute) ----
  mean_k<<<(BATCH * OUT_CH + 255) / 256, blk, 0, stream>>>(ob, mb);
  gemm_wmma<1><<<dim3(OUT_CH / 64, BATCH / 64), 256, 0, stream>>>(
      mb, opw, opb, out, BATCH, OUT_CH, OUT_CH, 0);
}
